// LSTM_Block_36258113913000
// MI455X (gfx1250) — compile-verified, matrix operand order checked by reading, not run on hardware
//
#include <hip/hip_runtime.h>
#include <hip/hip_bf16.h>

// ---------------------------------------------------------------------------
// LSTM block for MI455X (gfx1250, wave32, WMMA), round 2.
//   Kernel 1: fused 4-gate GEMM [8192x3072]x[2048x3072]^T + LSTM cell update
//   Kernel 2: out = tanh(C_new @ W_h2o^T + b) * o   [8192x2048]x[2048x2048]^T
// bf16 WMMA (v_wmma_f32_16x16x32_bf16), fp32 accumulate.
// Double-buffered LDS, one barrier per K-step, 16 WMMAs/wave/K-step.
// ---------------------------------------------------------------------------

typedef __attribute__((ext_vector_type(16))) __bf16 v16bf;
typedef __attribute__((ext_vector_type(8)))  __bf16 v8bf;
typedef __attribute__((ext_vector_type(8)))  float  v8f;

#define BATCH  8192
#define EMBED  1024
#define HIDDEN 2048
#define DIM    (EMBED + HIDDEN)   // 3072

#define BM  128                   // batch rows per block
#define BN  64                    // hidden cols per block
#define BK  32                    // K per step (bf16 WMMA K=32)
#define LDT 40                    // LDS pitch: 80B = 5*16B, aligned, conflict-free

__device__ __forceinline__ float sigmoid_f(float v) {
    return 1.0f / (1.0f + __expf(-v));
}
__device__ __forceinline__ float tanh_f(float v) {
    float e = __expf(2.0f * v);          // saturates to +/-1, no inf/inf NaN
    return 1.0f - 2.0f / (e + 1.0f);
}

// Pack 8 fp32 -> 8 bf16, one 16B LDS store (ds_store_b128).
__device__ __forceinline__ void stage8(__bf16* dst, const float* src) {
    float4 a = ((const float4*)src)[0];
    float4 b = ((const float4*)src)[1];
    v8bf p;
    p[0] = (__bf16)a.x; p[1] = (__bf16)a.y; p[2] = (__bf16)a.z; p[3] = (__bf16)a.w;
    p[4] = (__bf16)b.x; p[5] = (__bf16)b.y; p[6] = (__bf16)b.z; p[7] = (__bf16)b.w;
    *(v8bf*)dst = p;
}

// 16x32 bf16 WMMA operand fragment from an LDS tile row (CDNA5 ISA):
// lanes 0-15: row=lane,    K{0..7,16..23}; lanes 16-31: row=lane-16, K{8..15,24..31}.
// rowp = &tile[rowBase + (lane&15)][(lane>=16)*8]
__device__ __forceinline__ v16bf load_frag(const __bf16* rowp) {
    v8bf lo = *(const v8bf*)rowp;
    v8bf hi = *(const v8bf*)(rowp + 16);
    return __builtin_shufflevector(lo, hi, 0,1,2,3,4,5,6,7,8,9,10,11,12,13,14,15);
}

// ---------------------------------------------------------------------------
// Kernel 1: fused 4-gate GEMM + LSTM cell update.
// ---------------------------------------------------------------------------
__global__ __launch_bounds__(256)
void lstm_gates_kernel(const float* __restrict__ x,
                       const float* __restrict__ st,
                       const float* __restrict__ lt,
                       const float* __restrict__ Wf, const float* __restrict__ Bf,
                       const float* __restrict__ Wi, const float* __restrict__ Bi,
                       const float* __restrict__ Wc, const float* __restrict__ Bc,
                       const float* __restrict__ Wo, const float* __restrict__ Bo,
                       float* __restrict__ o_tmp,
                       float* __restrict__ c_new)
{
    __shared__ __bf16 sA[2][BM][LDT];        // 2*128*40*2 = 20 KB
    __shared__ __bf16 sB[2][4][BN][LDT];     // 2*4*64*40*2 = 40 KB

    const int tid  = threadIdx.x;
    const int lane = tid & 31;
    const int wave = tid >> 5;          // 0..7
    const int wm   = wave >> 2;         // 0..1 -> rows wm*64 .. +63 (4 subtiles)
    const int wn   = wave & 3;          // 0..3 -> cols wn*16
    const int half = lane >> 4;
    const int r    = lane & 15;

    const int m0 = blockIdx.y * BM;
    const int n0 = blockIdx.x * BN;

    // staging assignments
    const int arow = tid >> 1;          // 0..127 (A: 128 rows x 32 K)
    const int acol = (tid & 1) * 16;    // 0 / 16
    const int brow = tid >> 2;          // 0..63  (B: 64 rows x 32 K, x4 gates)
    const int bcol = (tid & 3) * 8;     // 0,8,16,24

    const float* W[4] = { Wf, Wi, Wc, Wo };

    v8f acc[4][4] = {};                 // [gate][m-subtile]

    auto stageA = [&](int buf, int k0) {
        const float* aptr;
        if (k0 < EMBED)
            aptr = x  + (size_t)(m0 + arow) * EMBED  + (k0 + acol);
        else
            aptr = st + (size_t)(m0 + arow) * HIDDEN + (k0 - EMBED + acol);
        stage8(&sA[buf][arow][acol],     aptr);
        stage8(&sA[buf][arow][acol + 8], aptr + 8);
        if (k0 + BK < DIM)
            __builtin_prefetch(aptr + BK, 0, 0);   // global_prefetch_b8
    };
    auto stageB = [&](int buf, int k0) {
#pragma unroll
        for (int g = 0; g < 4; ++g) {
            const float* wp = W[g] + (size_t)(n0 + brow) * DIM + (k0 + bcol);
            stage8(&sB[buf][g][brow][bcol], wp);
        }
    };

    stageA(0, 0);
    stageB(0, 0);
    __syncthreads();

    const int NT = DIM / BK;            // 96 K-steps
    for (int kt = 0; kt < NT; ++kt) {
        const int cur = kt & 1;
        // Prefetch next tile into the other buffer (overlaps the WMMA burst).
        if (kt + 1 < NT) {
            stageA(cur ^ 1, (kt + 1) * BK);
            stageB(cur ^ 1, (kt + 1) * BK);
        }

        v16bf afrag[4];
#pragma unroll
        for (int tm = 0; tm < 4; ++tm)
            afrag[tm] = load_frag(&sA[cur][wm * 64 + tm * 16 + r][half * 8]);

#pragma unroll
        for (int g = 0; g < 4; ++g) {
            v16bf bfrag = load_frag(&sB[cur][g][wn * 16 + r][half * 8]);
#pragma unroll
            for (int tm = 0; tm < 4; ++tm)
                acc[g][tm] = __builtin_amdgcn_wmma_f32_16x16x32_bf16(
                    false, afrag[tm], false, bfrag,
                    (short)0, acc[g][tm], false, false);
        }
        __syncthreads();                // single barrier per K-step
    }

    // ---- epilogue: gate activations + cell update (in-register fusion) ----
    const int nn = n0 + wn * 16 + r;                // C/D: lane(0..15) = N col
    const float bfv = Bf[nn], biv = Bi[nn], bcv = Bc[nn], bov = Bo[nn];

#pragma unroll
    for (int tm = 0; tm < 4; ++tm) {
        const int mbase = m0 + wm * 64 + tm * 16 + half * 8;
#pragma unroll
        for (int e = 0; e < 8; ++e) {
            const size_t idx = (size_t)(mbase + e) * HIDDEN + nn;
            float fv = sigmoid_f(acc[0][tm][e] + bfv);
            float iv = sigmoid_f(acc[1][tm][e] + biv);
            float cv = tanh_f   (acc[2][tm][e] + bcv);
            float ov = sigmoid_f(acc[3][tm][e] + bov);
            float cn = fv * lt[idx] + iv * cv;
            c_new[idx] = cn;            // second output (C_new)
            o_tmp[idx] = ov;            // temp: output gate, read by kernel 2
        }
    }
}

// ---------------------------------------------------------------------------
// Kernel 2: output = tanh(C_new @ W_h2o^T + b_h2o) * o  (o read from `out`,
// same element overwritten by the same thread -> safe in-place).
// ---------------------------------------------------------------------------
__global__ __launch_bounds__(256)
void lstm_out_kernel(const float* __restrict__ cnew,
                     const float* __restrict__ Wh,
                     const float* __restrict__ Bh,
                     float* __restrict__ out)
{
    __shared__ __bf16 sA[2][BM][LDT];   // 20 KB
    __shared__ __bf16 sB[2][BN][LDT];   // 10 KB

    const int tid  = threadIdx.x;
    const int lane = tid & 31;
    const int wave = tid >> 5;
    const int wm   = wave >> 2;
    const int wn   = wave & 3;
    const int half = lane >> 4;
    const int r    = lane & 15;

    const int m0 = blockIdx.y * BM;
    const int n0 = blockIdx.x * BN;

    const int arow = tid >> 1;
    const int acol = (tid & 1) * 16;
    const int brow = tid >> 2;
    const int bcol = (tid & 3) * 8;

    v8f acc[4] = {};

    auto stageA = [&](int buf, int k0) {
        const float* aptr = cnew + (size_t)(m0 + arow) * HIDDEN + (k0 + acol);
        stage8(&sA[buf][arow][acol],     aptr);
        stage8(&sA[buf][arow][acol + 8], aptr + 8);
        if (k0 + BK < HIDDEN)
            __builtin_prefetch(aptr + BK, 0, 0);
    };
    auto stageB = [&](int buf, int k0) {
        const float* wp = Wh + (size_t)(n0 + brow) * HIDDEN + (k0 + bcol);
        stage8(&sB[buf][brow][bcol], wp);
    };

    stageA(0, 0);
    stageB(0, 0);
    __syncthreads();

    const int NT = HIDDEN / BK;         // 64 K-steps
    for (int kt = 0; kt < NT; ++kt) {
        const int cur = kt & 1;
        if (kt + 1 < NT) {
            stageA(cur ^ 1, (kt + 1) * BK);
            stageB(cur ^ 1, (kt + 1) * BK);
        }

        v16bf bfrag = load_frag(&sB[cur][wn * 16 + r][half * 8]);
#pragma unroll
        for (int tm = 0; tm < 4; ++tm) {
            v16bf afrag = load_frag(&sA[cur][wm * 64 + tm * 16 + r][half * 8]);
            acc[tm] = __builtin_amdgcn_wmma_f32_16x16x32_bf16(
                false, afrag, false, bfrag, (short)0, acc[tm], false, false);
        }
        __syncthreads();
    }

    const int nn = n0 + wn * 16 + r;
    const float bv = Bh[nn];
#pragma unroll
    for (int tm = 0; tm < 4; ++tm) {
        const int mbase = m0 + wm * 64 + tm * 16 + half * 8;
#pragma unroll
        for (int e = 0; e < 8; ++e) {
            const size_t idx = (size_t)(mbase + e) * HIDDEN + nn;
            float ov = out[idx];                        // o from kernel 1
            out[idx] = tanh_f(acc[tm][e] + bv) * ov;    // final output
        }
    }
}

// ---------------------------------------------------------------------------
extern "C" void kernel_launch(void* const* d_in, const int* in_sizes, int n_in,
                              void* d_out, int out_size, void* d_ws, size_t ws_size,
                              hipStream_t stream) {
    (void)in_sizes; (void)n_in; (void)out_size; (void)d_ws; (void)ws_size;

    const float* x  = (const float*)d_in[0];
    const float* st = (const float*)d_in[1];
    const float* lt = (const float*)d_in[2];
    const float* Wf = (const float*)d_in[3];
    const float* Bf = (const float*)d_in[4];
    const float* Wi = (const float*)d_in[5];
    const float* Bi = (const float*)d_in[6];
    const float* Wc = (const float*)d_in[7];
    const float* Bc = (const float*)d_in[8];
    const float* Wo = (const float*)d_in[9];
    const float* Bo = (const float*)d_in[10];
    const float* Wh = (const float*)d_in[11];
    const float* Bh = (const float*)d_in[12];

    float* out   = (float*)d_out;                    // output [B, H]
    float* c_new = out + (size_t)BATCH * HIDDEN;     // C_new  [B, H]

    dim3 blk(256);
    dim3 grd(HIDDEN / BN, BATCH / BM);               // (32, 64)

    lstm_gates_kernel<<<grd, blk, 0, stream>>>(x, st, lt,
                                               Wf, Bf, Wi, Bi, Wc, Bc, Wo, Bo,
                                               out, c_new);
    lstm_out_kernel<<<grd, blk, 0, stream>>>(c_new, Wh, Bh, out);
}
